// SymmetricHyperRBM_33767032882057
// MI455X (gfx1250) — compile-verified, matrix operand order checked by reading, not run on hardware
//
#include <hip/hip_runtime.h>
#include <stdint.h>

// ---------------------------------------------------------------------------
// SymmetricHyperRBM on gfx1250 (MI455X): WMMA bf16 GEMM pipeline with
// async global->LDS staging (ASYNCcnt) and double-buffered K-tiles.
//   sizes: B=16384, NUM_V=NUM_H=1024, HYPER_DIM=64, K=10 Gibbs steps, T=1.
// ---------------------------------------------------------------------------

#define BATCH   16384
#define NV      1024
#define NH      1024
#define HD      64
#define KSTEPS  10

typedef __bf16 bf16;
typedef bf16  v8bf  __attribute__((ext_vector_type(8)));
typedef bf16  v16bf __attribute__((ext_vector_type(16)));
typedef float v8f   __attribute__((ext_vector_type(8)));

#define BFCAT(lo, hi) __builtin_shufflevector((lo), (hi), 0,1,2,3,4,5,6,7,8,9,10,11,12,13,14,15)
#define WMMA_BF16(a, b, c) \
  __builtin_amdgcn_wmma_f32_16x16x32_bf16(false, (a), false, (b), (short)0, (c), false, false)

// ---- async global->LDS path (guarded; falls back to plain staging) --------
#if __has_builtin(__builtin_amdgcn_global_load_async_to_lds_b128) && \
    __has_builtin(__builtin_amdgcn_s_wait_asynccnt)
#define HAS_ASYNC_LDS 1
#endif

typedef int v4i __attribute__((__vector_size__(4 * sizeof(int))));
typedef __attribute__((address_space(1))) v4i gas_v4i;   // global (prints as __device__)
typedef __attribute__((address_space(3))) v4i las_v4i;   // LDS    (prints as __shared__)

__device__ __forceinline__ void cp16_g2l(const bf16* g, bf16* l) {
#ifdef HAS_ASYNC_LDS
  __builtin_amdgcn_global_load_async_to_lds_b128((gas_v4i*)g, (las_v4i*)l, 0, 0);
#else
  *(v8bf*)l = *(const v8bf*)g;
#endif
}

__device__ __forceinline__ void wait_async_lds() {
#ifdef HAS_ASYNC_LDS
  __builtin_amdgcn_s_wait_asynccnt(0);
#endif
}

// ---------------- small device helpers ----------------

__device__ __forceinline__ float rnd01(unsigned s, unsigned i) {
  unsigned x = i + s * 0x9E3779B9u;
  x ^= x >> 16; x *= 0x7FEB352Du;
  x ^= x >> 15; x *= 0x846CA68Bu;
  x ^= x >> 16;
  return (float)(x >> 8) * (1.0f / 16777216.0f);
}

__device__ __forceinline__ float sigmoidf(float x) { return 1.0f / (1.0f + expf(-x)); }

__device__ __forceinline__ float softplusf(float x) {
  return (x > 20.0f) ? x : log1pf(expf(x));
}

__device__ __forceinline__ float wave_red(float v) {
#pragma unroll
  for (int o = 16; o > 0; o >>= 1) v += __shfl_down(v, o, 32);
  return v;
}

// ---------------- setup kernels ----------------

// W f32 [NV][NH] -> Wbf (row-major) and Wtbf (transposed, [NH][NV])
__global__ void __launch_bounds__(256) k_convert_W(const float* __restrict__ W,
                                                   bf16* __restrict__ Wbf,
                                                   bf16* __restrict__ Wtbf) {
  int idx = blockIdx.x * 256 + threadIdx.x;           // < NV*NH
  int i = idx >> 10, j = idx & 1023;
  float w = W[idx];
  Wbf[idx] = (bf16)w;
  Wtbf[j * NV + i] = (bf16)w;
}

// colsum[n] = sum_i W[i][n]
__global__ void __launch_bounds__(256) k_colsum(const float* __restrict__ W,
                                                float* __restrict__ colsum) {
  int n = blockIdx.x * 256 + threadIdx.x;             // < NH
  float s = 0.0f;
  for (int i = 0; i < NV; ++i) s += W[i * NH + n];
  colsum[n] = s;
}

__global__ void __launch_bounds__(256) k_convert_fc2(const float* __restrict__ fc2_w,
                                                     bf16* __restrict__ fc2bf) {
  int idx = blockIdx.x * 256 + threadIdx.x;           // < 4096*64
  fc2bf[idx] = (bf16)fc2_w[idx];
}

// x[i][d] = tanh(cond[i]*fc1_w[d] + fc1_b[d])  (bf16)
__global__ void __launch_bounds__(256) k_hyper_x(const float* __restrict__ cond,
                                                 const float* __restrict__ fc1_w,
                                                 const float* __restrict__ fc1_b,
                                                 bf16* __restrict__ xbf) {
  int idx = blockIdx.x * 256 + threadIdx.x;           // < BATCH*HD
  int i = idx >> 6, d = idx & 63;
  xbf[idx] = (bf16)tanhf(cond[i] * fc1_w[d] + fc1_b[d]);
}

// v_data f32 -> v_cur bf16 and vdata_bf bf16
__global__ void __launch_bounds__(256) k_convert_v(const float* __restrict__ vd,
                                                   bf16* __restrict__ v_cur,
                                                   bf16* __restrict__ vdata_bf) {
  int idx = blockIdx.x * 256 + threadIdx.x;
  float v = vd[idx];
  v_cur[idx] = (bf16)v;
  vdata_bf[idx] = (bf16)v;
}

__global__ void k_zero_loss(float* loss) { if (threadIdx.x == 0) loss[0] = 0.0f; }

__global__ void __launch_bounds__(256) k_u0(float* __restrict__ u_buf, unsigned seed) {
  int m = blockIdx.x * 256 + threadIdx.x;             // < BATCH
  u_buf[m] = (rnd01(seed, (unsigned)m) < 0.5f) ? 1.0f : 0.0f;
}

// ---------------- hypernet bias-mod GEMM (WMMA, K=64) ----------------
// y = x @ fc2_w^T ; b_mod = (1+gamma)*base + beta, fused per 16x16 tile.
// grid: (NV/16, BATCH/128, 2)  z=0 -> b_mod, z=1 -> c_mod
__global__ void __launch_bounds__(256) k_biasmod(const bf16* __restrict__ xbf,
                                                 const bf16* __restrict__ fc2bf,
                                                 const float* __restrict__ fc2_b,
                                                 const float* __restrict__ bvec,
                                                 const float* __restrict__ cvec,
                                                 float* __restrict__ bmod,
                                                 float* __restrict__ cmod) {
  const int tid = threadIdx.x, lane = tid & 31, wv = tid >> 5;
  const int hf = lane >> 4, l16 = lane & 15;
  const int n0 = blockIdx.x * 16;
  const int m0 = blockIdx.y * 128 + wv * 16;
  const int z = blockIdx.z;
  const float* base = z ? cvec : bvec;
  float* outp = z ? cmod : bmod;
  const int jg0 = z * 2048 + n0;        // gamma rows of fc2_w
  const int jb0 = jg0 + 1024;           // beta rows

  v8f ag = {}, ab = {};
#pragma unroll
  for (int k0 = 0; k0 < HD; k0 += 32) {
    const int m = m0 + l16;
    v8bf x0 = *(const v8bf*)&xbf[(size_t)m * HD + k0 + hf * 8];
    v8bf x1 = *(const v8bf*)&xbf[(size_t)m * HD + k0 + 16 + hf * 8];
    v16bf af = BFCAT(x0, x1);
    const int jg = jg0 + l16, jb = jb0 + l16;
    v8bf g0 = *(const v8bf*)&fc2bf[(size_t)jg * HD + k0 + hf * 8];
    v8bf g1 = *(const v8bf*)&fc2bf[(size_t)jg * HD + k0 + 16 + hf * 8];
    v16bf bg = BFCAT(g0, g1);
    v8bf b0 = *(const v8bf*)&fc2bf[(size_t)jb * HD + k0 + hf * 8];
    v8bf b1 = *(const v8bf*)&fc2bf[(size_t)jb * HD + k0 + 16 + hf * 8];
    v16bf bb = BFCAT(b0, b1);
    ag = WMMA_BF16(af, bg, ag);
    ab = WMMA_BF16(af, bb, ab);
  }
  const int n = n0 + l16;
  const float biasg = fc2_b[z * 2048 + n];
  const float biasb = fc2_b[z * 2048 + 1024 + n];
  const float bs = base[n];
#pragma unroll
  for (int r = 0; r < 8; ++r) {
    int m = m0 + r + hf * 8;
    float g = ag[r] + biasg;
    float bt = ab[r] + biasb;
    outp[(size_t)m * 1024 + n] = (1.0f + g) * bs + bt;
  }
}

// ---------------- main tiled WMMA GEMM  (M=16384, N=1024, K=1024) ----------------
// C[m][n] = sum_k A[m][k] * Bn[n][k]   (Bn is the N x K operand, row-major)
// Double-buffered LDS with async global->LDS staging (ASYNCcnt).
// EPI 0: store f32 to outF.   EPI 1: h = bernoulli(sigmoid(acc + cmod)) -> outH bf16.
#define LDSS 40   // padded LDS row stride (bf16 elems): 80B step -> conflict-free

template <int EPI>
__global__ void __launch_bounds__(256) gemm_k(const bf16* __restrict__ A,
                                              const bf16* __restrict__ Bn,
                                              const float* __restrict__ cmod,
                                              float* __restrict__ outF,
                                              bf16* __restrict__ outH,
                                              unsigned seed) {
  __shared__ bf16 sA[2][128 * LDSS];
  __shared__ bf16 sB[2][128 * LDSS];
  const int tid = threadIdx.x, lane = tid & 31, wv = tid >> 5;
  const int wm = wv & 3, wn = wv >> 2;
  const int Mb = blockIdx.y * 128, Nb = blockIdx.x * 128;
  const int hf = lane >> 4, l16 = lane & 15;

  // staging assignment: 512 16B chunks per tile; this thread owns 2 of each.
  const int row0 = tid >> 2, off0 = (tid & 3) * 8;             // chunk tid
  const int row1 = (tid + 256) >> 2, off1 = ((tid + 256) & 3) * 8;

  auto stage = [&](int kt, int p) {
    cp16_g2l(&A[(size_t)(Mb + row0) * 1024 + kt + off0], &sA[p][row0 * LDSS + off0]);
    cp16_g2l(&Bn[(size_t)(Nb + row0) * 1024 + kt + off0], &sB[p][row0 * LDSS + off0]);
    cp16_g2l(&A[(size_t)(Mb + row1) * 1024 + kt + off1], &sA[p][row1 * LDSS + off1]);
    cp16_g2l(&Bn[(size_t)(Nb + row1) * 1024 + kt + off1], &sB[p][row1 * LDSS + off1]);
  };

  v8f acc[2][4] = {};

  stage(0, 0);
  wait_async_lds();
  __syncthreads();

  int p = 0;
  for (int kt = 0; kt < 1024; kt += 32) {
    if (kt + 32 < 1024) stage(kt + 32, p ^ 1);   // overlap next tile with compute

    v16bf af[2], bfr[4];
#pragma unroll
    for (int mf = 0; mf < 2; ++mf) {
      int r = wm * 32 + mf * 16 + l16;
      v8bf x0 = *(const v8bf*)&sA[p][r * LDSS + hf * 8];
      v8bf x1 = *(const v8bf*)&sA[p][r * LDSS + 16 + hf * 8];
      af[mf] = BFCAT(x0, x1);
    }
#pragma unroll
    for (int nf = 0; nf < 4; ++nf) {
      int r = wn * 64 + nf * 16 + l16;
      v8bf x0 = *(const v8bf*)&sB[p][r * LDSS + hf * 8];
      v8bf x1 = *(const v8bf*)&sB[p][r * LDSS + 16 + hf * 8];
      bfr[nf] = BFCAT(x0, x1);
    }
#pragma unroll
    for (int mf = 0; mf < 2; ++mf)
#pragma unroll
      for (int nf = 0; nf < 4; ++nf)
        acc[mf][nf] = WMMA_BF16(af[mf], bfr[nf], acc[mf][nf]);

    wait_async_lds();
    __syncthreads();
    p ^= 1;
  }

  // epilogue
#pragma unroll
  for (int mf = 0; mf < 2; ++mf)
#pragma unroll
    for (int nf = 0; nf < 4; ++nf)
#pragma unroll
      for (int r = 0; r < 8; ++r) {
        int m = Mb + wm * 32 + mf * 16 + r + hf * 8;
        int n = Nb + wn * 64 + nf * 16 + l16;
        size_t idx = (size_t)m * 1024 + n;
        float v = acc[mf][nf][r];
        if (EPI == 0) {
          outF[idx] = v;
        } else {
          float pr = sigmoidf(v + cmod[idx]);                     // T = 1
          outH[idx] = (bf16)((rnd01(seed, (unsigned)idx) < pr) ? 1.0f : 0.0f);
        }
      }
}

// ---------------- Gibbs elementwise / reduction kernels ----------------

// v_branch = u*v + (1-u)*(1-v)
__global__ void __launch_bounds__(256) k_vbranch(const bf16* __restrict__ v_cur,
                                                 const float* __restrict__ u_buf,
                                                 bf16* __restrict__ vbr) {
  int idx = blockIdx.x * 256 + threadIdx.x;
  int m = idx >> 10;
  float u = u_buf[m], v = (float)v_cur[idx];
  vbr[idx] = (bf16)(u * v + (1.0f - u) * (1.0f - v));
}

// per-row: dE = -sum(bmod) - sum(a) + 2*sum(v*bmod) + 2*sum(v*a);  u ~ Bern(sigmoid(dE))
__global__ void __launch_bounds__(256) k_dE_u(const float* __restrict__ a_buf,
                                              const float* __restrict__ bmod,
                                              const bf16* __restrict__ v_cur,
                                              float* __restrict__ u_buf,
                                              unsigned seed) {
  __shared__ float sm[32];
  const int m = blockIdx.x, tid = threadIdx.x, lane = tid & 31, wv = tid >> 5;
  float sa = 0, sva = 0, sb = 0, svb = 0;
  for (int n = tid; n < 1024; n += 256) {
    size_t idx = (size_t)m * 1024 + n;
    float av = a_buf[idx], bm = bmod[idx], vv = (float)v_cur[idx];
    sa += av; sva += vv * av; sb += bm; svb += vv * bm;
  }
  sa = wave_red(sa); sva = wave_red(sva); sb = wave_red(sb); svb = wave_red(svb);
  if (lane == 0) { sm[wv] = sa; sm[8 + wv] = sva; sm[16 + wv] = sb; sm[24 + wv] = svb; }
  __syncthreads();
  if (tid == 0) {
    float S0 = 0, S1 = 0, S2 = 0, S3 = 0;
    for (int i = 0; i < 8; ++i) { S0 += sm[i]; S1 += sm[8 + i]; S2 += sm[16 + i]; S3 += sm[24 + i]; }
    float dE = -S2 - S0 + 2.0f * S3 + 2.0f * S1;
    float pr = sigmoidf(dE);
    u_buf[m] = (rnd01(seed, (unsigned)m) < pr) ? 1.0f : 0.0f;
  }
}

// p_v = sigmoid(a + bmod); v_next = u*vb + (1-u)*(1-vb)
__global__ void __launch_bounds__(256) k_update_v(const float* __restrict__ a_buf,
                                                  const float* __restrict__ bmod,
                                                  const float* __restrict__ u_buf,
                                                  bf16* __restrict__ v_cur,
                                                  unsigned seed) {
  int idx = blockIdx.x * 256 + threadIdx.x;
  int m = idx >> 10;
  float pr = sigmoidf(a_buf[idx] + bmod[idx]);
  float vb = (rnd01(seed, (unsigned)idx) < pr) ? 1.0f : 0.0f;
  float u = u_buf[m];
  v_cur[idx] = (bf16)(u * vb + (1.0f - u) * (1.0f - vb));
}

// free-energy per-row reduction; loss += sign * F(row) / BATCH
__global__ void __launch_bounds__(256) k_fe(const float* __restrict__ vW,
                                            const float* __restrict__ cmod,
                                            const float* __restrict__ bmod,
                                            const bf16* __restrict__ v,
                                            const float* __restrict__ colsum,
                                            float sign, float* __restrict__ loss) {
  __shared__ float sm[32];
  const int m = blockIdx.x, tid = threadIdx.x, lane = tid & 31, wv = tid >> 5;
  float sp_n = 0, sp_f = 0, svb = 0, sb = 0;
  for (int n = tid; n < 1024; n += 256) {
    size_t idx = (size_t)m * 1024 + n;
    float t = vW[idx], cm = cmod[idx], bm = bmod[idx], vv = (float)v[idx];
    sp_n += softplusf(t + cm);
    sp_f += softplusf(colsum[n] - t + cm);
    svb += vv * bm;
    sb += bm;
  }
  sp_n = wave_red(sp_n); sp_f = wave_red(sp_f); svb = wave_red(svb); sb = wave_red(sb);
  if (lane == 0) { sm[wv] = sp_n; sm[8 + wv] = sp_f; sm[16 + wv] = svb; sm[24 + wv] = sb; }
  __syncthreads();
  if (tid == 0) {
    float A1 = 0, A2 = 0, A3 = 0, A4 = 0;
    for (int i = 0; i < 8; ++i) { A1 += sm[i]; A2 += sm[8 + i]; A3 += sm[16 + i]; A4 += sm[24 + i]; }
    float Fn = -A3 - A1;                    // vb_normal - sp_normal
    float Ff = -(A4 - A3) - A2;             // vb_flipped - sp_flipped
    float a = -Fn, b = -Ff;
    float mx = fmaxf(a, b);
    float lse = mx + logf(expf(a - mx) + expf(b - mx));
    float F = -lse;                         // T = 1
    atomicAdd(loss, sign * F * (1.0f / (float)BATCH));
  }
}

// ---------------------------------------------------------------------------

extern "C" void kernel_launch(void* const* d_in, const int* in_sizes, int n_in,
                              void* d_out, int out_size, void* d_ws, size_t ws_size,
                              hipStream_t stream) {
  const float* v_data = (const float*)d_in[0];
  const float* cond   = (const float*)d_in[1];
  const float* W      = (const float*)d_in[2];
  const float* bvec   = (const float*)d_in[3];
  const float* cvec   = (const float*)d_in[4];
  const float* fc1_w  = (const float*)d_in[5];
  const float* fc1_b  = (const float*)d_in[6];
  const float* fc2_w  = (const float*)d_in[7];
  const float* fc2_b  = (const float*)d_in[8];
  float* loss = (float*)d_out;

  char* w = (char*)d_ws;
  auto alloc = [&](size_t bytes) -> char* {
    char* p = w;
    w += (bytes + 255) & ~(size_t)255;
    return p;
  };
  bf16*  Wbf     = (bf16*)alloc((size_t)NV * NH * 2);
  bf16*  Wtbf    = (bf16*)alloc((size_t)NV * NH * 2);
  bf16*  fc2bf   = (bf16*)alloc((size_t)4096 * HD * 2);
  bf16*  xbf     = (bf16*)alloc((size_t)BATCH * HD * 2);
  float* colsum  = (float*)alloc((size_t)NH * 4);
  float* b_mod   = (float*)alloc((size_t)BATCH * NV * 4);
  float* c_mod   = (float*)alloc((size_t)BATCH * NH * 4);
  bf16*  v_cur   = (bf16*)alloc((size_t)BATCH * NV * 2);
  bf16*  vdatabf = (bf16*)alloc((size_t)BATCH * NV * 2);
  bf16*  vbr     = (bf16*)alloc((size_t)BATCH * NV * 2);
  bf16*  h_bf    = (bf16*)alloc((size_t)BATCH * NH * 2);
  float* a_buf   = (float*)alloc((size_t)BATCH * NV * 4);
  float* u_buf   = (float*)alloc((size_t)BATCH * 4);
  (void)ws_size; (void)in_sizes; (void)n_in; (void)out_size;

  const dim3 blk(256);
  const dim3 gemm_grid(NH / 128, BATCH / 128);   // (8, 128)
  const int elem_blocks = (BATCH * NV) / 256;    // 65536

  // ---- setup ----
  k_convert_W  <<<(NV * NH) / 256, blk, 0, stream>>>(W, Wbf, Wtbf);
  k_colsum     <<<NH / 256, blk, 0, stream>>>(W, colsum);
  k_convert_fc2<<<(4096 * HD) / 256, blk, 0, stream>>>(fc2_w, fc2bf);
  k_hyper_x    <<<(BATCH * HD) / 256, blk, 0, stream>>>(cond, fc1_w, fc1_b, xbf);
  k_biasmod    <<<dim3(NV / 16, BATCH / 128, 2), blk, 0, stream>>>(
      xbf, fc2bf, fc2_b, bvec, cvec, b_mod, c_mod);
  k_convert_v  <<<elem_blocks, blk, 0, stream>>>(v_data, v_cur, vdatabf);
  k_zero_loss  <<<1, 32, 0, stream>>>(loss);
  k_u0         <<<BATCH / 256, blk, 0, stream>>>(u_buf, 11u);

  // ---- Gibbs chain ----
  for (int s = 0; s < KSTEPS; ++s) {
    unsigned hs = 100u + (unsigned)s * 7u;
    k_vbranch<<<elem_blocks, blk, 0, stream>>>(v_cur, u_buf, vbr);
    // p_h = sigmoid(v_branch @ W + c_mod) ; h ~ Bern(p_h)
    gemm_k<1><<<gemm_grid, blk, 0, stream>>>(vbr, Wtbf, c_mod, nullptr, h_bf, hs);
    // a = h @ W^T
    gemm_k<0><<<gemm_grid, blk, 0, stream>>>(h_bf, Wbf, nullptr, a_buf, nullptr, 0u);
    k_dE_u    <<<BATCH, blk, 0, stream>>>(a_buf, b_mod, v_cur, u_buf, hs + 1u);
    k_update_v<<<elem_blocks, blk, 0, stream>>>(a_buf, b_mod, u_buf, v_cur, hs + 2u);
  }

  // ---- free energy: loss = mean F(v_data) - mean F(v_model) ----
  gemm_k<0><<<gemm_grid, blk, 0, stream>>>(vdatabf, Wtbf, nullptr, a_buf, nullptr, 0u);
  k_fe<<<BATCH, blk, 0, stream>>>(a_buf, c_mod, b_mod, vdatabf, colsum, +1.0f, loss);

  gemm_k<0><<<gemm_grid, blk, 0, stream>>>(v_cur, Wtbf, nullptr, a_buf, nullptr, 0u);
  k_fe<<<BATCH, blk, 0, stream>>>(a_buf, c_mod, b_mod, v_cur, colsum, -1.0f, loss);
}
